// GRU_25709674233937
// MI455X (gfx1250) — matrix operand order repaired and checked by reading, not compiled, not run on hardware
//
#include <hip/hip_runtime.h>
#include <hip/hip_bf16.h>
#include <math.h>

// ---------------- problem sizes ----------------
#define T_STEPS 256
#define BATCH   128
#define IN_DIM  1024
#define HID     1024
#define OUT_DIM 512
#define BH      (BATCH * HID)

// ---------------- WMMA types ----------------
typedef __attribute__((ext_vector_type(16))) __bf16 v16bf;
typedef __attribute__((ext_vector_type(8)))  float  v8f;
typedef unsigned int u32x4 __attribute__((ext_vector_type(4)));
typedef int          i32x8 __attribute__((ext_vector_type(8)));
typedef int          i32x4 __attribute__((ext_vector_type(4)));

#define KC 32            // K elements consumed per WMMA_16x16x32
#define NCH (HID / KC)   // K chunks in the recurrent GEMMs
#define LDSS 48          // padded LDS row stride in bf16 elems (96B, 16B-aligned subtiles)

#define WMMA_BF16(a, b, c) \
  __builtin_amdgcn_wmma_f32_16x16x32_bf16(false, (a), false, (b), (short)0, (c), false, false)

// ---------------- TDM availability ----------------
#if defined(__has_builtin)
#if __has_builtin(__builtin_amdgcn_tensor_load_to_lds) && \
    __has_builtin(__builtin_amdgcn_s_wait_tensorcnt)
#define GRU_USE_TDM 1
#if __has_include(<hip/amd_detail/amd_gfx1250_TDM.h>)
#define GRU_TDM_6ARG 1   // therock-10.0 headers -> clang-23 6-arg builtin
#endif
#endif
#endif
#ifndef GRU_USE_TDM
#define GRU_USE_TDM 0
#endif

#if GRU_USE_TDM
// 2-D TDM tile load: tile_rows x 32 bf16 elements from a row-major matrix with
// row stride HID elems, into LDS with a 96B row pitch (64B data + 32B pad via
// the TDM pad feature: pad_interval=64B (code 3), pad_amount=8 DWORDs (code 7)).
// D# layout per CDNA5 ISA 8.3/8.4.
__device__ __forceinline__ void tdm_load_2d(unsigned int lds_off, const void* gsrc,
                                            int tile_rows) {
  unsigned long long ga = (unsigned long long)(uintptr_t)gsrc;
  u32x4 g0;
  g0.x = 1u;                                   // count=1, user descriptor, no gather
  g0.y = lds_off;                              // lds_addr (bytes)
  g0.z = (unsigned int)ga;                     // global_addr[31:0]
  g0.w = (unsigned int)((ga >> 32) & 0x01FFFFFFu) | (2u << 30); // addr[56:32] | type=2
  i32x8 g1 = (i32x8)0;
  g1[0] = (int)((1u << 16)      // data_size = 2B
              | (1u << 20)      // pad_enable
              | (3u << 22)      // pad_interval: 16 DWORDs = 64B
              | (7u << 25));    // pad_amount: 8 DWORDs = 32B
  g1[1] = (int)(((unsigned)HID & 0xFFFFu) << 16);          // tensor_dim0[15:0] = 1024
  g1[2] = (int)(((unsigned)tile_rows & 0xFFFFu) << 16);    // tensor_dim1[15:0]
  g1[3] = (int)(((unsigned)KC & 0xFFFFu) << 16);           // tile_dim0 = 32
  g1[4] = tile_rows;                                       // tile_dim1
  g1[5] = HID;                                             // tensor_dim0_stride = 1024
  i32x4 z4 = (i32x4)0;
#if defined(GRU_TDM_6ARG)
  i32x8 z8 = (i32x8)0;
  __builtin_amdgcn_tensor_load_to_lds(g0, g1, z4, z4, z8, 0);
#else
  __builtin_amdgcn_tensor_load_to_lds(g0, g1, z4, z4, 0);
#endif
}
#endif  // GRU_USE_TDM

// A fragment: 16(M) x 32(K) bf16, per ISA 16-bit A layout.
__device__ __forceinline__ v16bf frag_a(const __bf16* lds, int lane, int m0) {
  const int m  = m0 + (lane & 15);
  const int kb = (lane >> 4) * 8;
  const __bf16* p = lds + m * LDSS + kb;
  v16bf f;
  ((uint4*)&f)[0] = *(const uint4*)(p);        // K = kb .. kb+7
  ((uint4*)&f)[1] = *(const uint4*)(p + 16);   // K = kb+16 .. kb+23
  return f;
}

// B fragment: 32(K) x 16(N) bf16 from row-major W[n][k];
// lanes 0-15: N=lane, K=0..7,16..23; lanes 16-31: N=lane&15, K=8..15,24..31.
__device__ __forceinline__ v16bf frag_b(const __bf16* lds, int lane, int n0) {
  const int n  = n0 + (lane & 15);
  const int kb = (lane >> 4) * 8;
  const __bf16* p = lds + n * LDSS + kb;
  v16bf f;
  ((uint4*)&f)[0] = *(const uint4*)(p);
  ((uint4*)&f)[1] = *(const uint4*)(p + 16);
  return f;
}

__device__ __forceinline__ float sigmoid_f(float x) { return 1.0f / (1.0f + __expf(-x)); }

// ---------------- weight fp32 -> bf16 ----------------
__global__ __launch_bounds__(256) void cvt_f32_bf16(const float* __restrict__ s,
                                                    __bf16* __restrict__ d, int n) {
  int i = (blockIdx.x * 256 + threadIdx.x) * 4;
  if (i + 3 < n) {
    float4 v = *(const float4*)(s + i);
    d[i + 0] = (__bf16)v.x; d[i + 1] = (__bf16)v.y;
    d[i + 2] = (__bf16)v.z; d[i + 3] = (__bf16)v.w;
  }
}

__global__ __launch_bounds__(256) void init_h_kernel(float* __restrict__ h,
                                                     __bf16* __restrict__ hb, int n) {
  int i = blockIdx.x * 256 + threadIdx.x;
  if (i < n) { h[i] = 0.0f; hb[i] = (__bf16)0.0f; }
}

// ---------------- P_g = (x + noise_g) @ Wx_g^T + b_g  (M=T*B, N=HID, K=IN_DIM) ----------
__global__ __launch_bounds__(256)
void input_proj_kernel(const float* __restrict__ x, const float* __restrict__ noise,
                       const __bf16* __restrict__ W, const float* __restrict__ bias,
                       float* __restrict__ P, int M, int N, int K) {
  __shared__ __bf16 ldsA[128 * LDSS];
  __shared__ __bf16 ldsB[64 * LDSS];
  const int tid = threadIdx.x, lane = tid & 31, wave = tid >> 5;
  const int wm = wave & 3, wn = wave >> 2;
  const int bm = blockIdx.y, bn = blockIdx.x;

  v8f acc[2][2] = {};
  const int am = tid >> 1, ah = (tid & 1) * 16;     // A fill: 16 bf16 per thread
  const int bnr = tid >> 2, bq = (tid & 3) * 8;     // B fill: 8 bf16 per thread
  const size_t arow = (size_t)(bm * 128 + am) * K + ah;
  const size_t brow = (size_t)(bn * 64 + bnr) * K + bq;

  for (int k0 = 0; k0 < K; k0 += KC) {
    // ---- stage A = bf16(x + noise) ----
    const float4* xp = (const float4*)(x + arow + k0);
    const float4* np = (const float4*)(noise + arow + k0);
    __bf16 tmp[16] __attribute__((aligned(16)));
#pragma unroll
    for (int v = 0; v < 4; ++v) {
      float4 xv = xp[v], nv = np[v];
      tmp[4 * v + 0] = (__bf16)(xv.x + nv.x);
      tmp[4 * v + 1] = (__bf16)(xv.y + nv.y);
      tmp[4 * v + 2] = (__bf16)(xv.z + nv.z);
      tmp[4 * v + 3] = (__bf16)(xv.w + nv.w);
    }
    __bf16* adst = &ldsA[am * LDSS + ah];
    ((uint4*)adst)[0] = ((const uint4*)tmp)[0];
    ((uint4*)adst)[1] = ((const uint4*)tmp)[1];
    // ---- stage B = W tile (already bf16) ----
    *(uint4*)&ldsB[bnr * LDSS + bq] = *(const uint4*)(W + brow + k0);
    if (k0 + KC < K) {
      __builtin_prefetch(x + arow + k0 + KC, 0, 1);
      __builtin_prefetch(noise + arow + k0 + KC, 0, 1);
    }
    __syncthreads();
    v16bf a0 = frag_a(ldsA, lane, wm * 32);
    v16bf a1 = frag_a(ldsA, lane, wm * 32 + 16);
    v16bf b0 = frag_b(ldsB, lane, wn * 32);
    v16bf b1 = frag_b(ldsB, lane, wn * 32 + 16);
    acc[0][0] = WMMA_BF16(a0, b0, acc[0][0]);
    acc[0][1] = WMMA_BF16(a0, b1, acc[0][1]);
    acc[1][0] = WMMA_BF16(a1, b0, acc[1][0]);
    acc[1][1] = WMMA_BF16(a1, b1, acc[1][1]);
    __syncthreads();
  }
  const int mloc = (lane >> 4) * 8, nloc = lane & 15;
#pragma unroll
  for (int ti = 0; ti < 2; ++ti)
#pragma unroll
    for (int tj = 0; tj < 2; ++tj) {
      int gn = bn * 64 + wn * 32 + tj * 16 + nloc;
      float bb = bias[gn];
#pragma unroll
      for (int j = 0; j < 8; ++j) {
        int gm = bm * 128 + wm * 32 + ti * 16 + mloc + j;
        P[(size_t)gm * N + gn] = acc[ti][tj][j] + bb;
      }
    }
}

// ---------------- per-step R/Z gates: dual GEMM on h (M=128, N=HID, K=HID) -------------
__global__ __launch_bounds__(256)
void gate_rz_kernel(const __bf16* __restrict__ hB, const __bf16* __restrict__ Whr,
                    const __bf16* __restrict__ Whz, const float* __restrict__ Pr,
                    const float* __restrict__ Pz, const float* __restrict__ h,
                    float* __restrict__ Zbuf, __bf16* __restrict__ RhB) {
  const int tid = threadIdx.x, lane = tid & 31, wave = tid >> 5;
  const int wm = wave & 3, wn = wave >> 2;
  const int bn = blockIdx.x;
  v8f aR[2][2] = {};
  v8f aZ[2][2] = {};

#if GRU_USE_TDM
  // Double-buffered LDS; wave 0 drives the Tensor Data Mover (3 tiles per chunk),
  // all waves run the WMMA chain.  TDM ops from one wave complete in order, so
  // s_wait_tensorcnt(3) == "previous chunk's group of 3 is done".
  __shared__ __bf16 ldsA[2][128 * LDSS];
  __shared__ __bf16 ldsBr[2][64 * LDSS];
  __shared__ __bf16 ldsBz[2][64 * LDSS];
  const __bf16* wrow_r = Whr + (size_t)(bn * 64) * HID;
  const __bf16* wrow_z = Whz + (size_t)(bn * 64) * HID;
  if (wave == 0) {
    tdm_load_2d((unsigned int)(uintptr_t)&ldsA[0][0],  hB,          128);
    tdm_load_2d((unsigned int)(uintptr_t)&ldsBr[0][0], wrow_r,      64);
    tdm_load_2d((unsigned int)(uintptr_t)&ldsBz[0][0], wrow_z,      64);
    tdm_load_2d((unsigned int)(uintptr_t)&ldsA[1][0],  hB + KC,     128);
    tdm_load_2d((unsigned int)(uintptr_t)&ldsBr[1][0], wrow_r + KC, 64);
    tdm_load_2d((unsigned int)(uintptr_t)&ldsBz[1][0], wrow_z + KC, 64);
  }
  for (int i = 0; i < NCH; ++i) {
    if (i < NCH - 1) __builtin_amdgcn_s_wait_tensorcnt(3);   // chunk i landed
    else             __builtin_amdgcn_s_wait_tensorcnt(0);
    __syncthreads();
    const int buf = i & 1;
    v16bf a0 = frag_a(ldsA[buf], lane, wm * 32);
    v16bf a1 = frag_a(ldsA[buf], lane, wm * 32 + 16);
    v16bf r0 = frag_b(ldsBr[buf], lane, wn * 32);
    v16bf r1 = frag_b(ldsBr[buf], lane, wn * 32 + 16);
    v16bf z0 = frag_b(ldsBz[buf], lane, wn * 32);
    v16bf z1 = frag_b(ldsBz[buf], lane, wn * 32 + 16);
    aR[0][0] = WMMA_BF16(a0, r0, aR[0][0]);
    aR[0][1] = WMMA_BF16(a0, r1, aR[0][1]);
    aR[1][0] = WMMA_BF16(a1, r0, aR[1][0]);
    aR[1][1] = WMMA_BF16(a1, r1, aR[1][1]);
    aZ[0][0] = WMMA_BF16(a0, z0, aZ[0][0]);
    aZ[0][1] = WMMA_BF16(a0, z1, aZ[0][1]);
    aZ[1][0] = WMMA_BF16(a1, z0, aZ[1][0]);
    aZ[1][1] = WMMA_BF16(a1, z1, aZ[1][1]);
    __syncthreads();                                          // buf free for reuse
    if (wave == 0 && i + 2 < NCH) {
      const int k2 = (i + 2) * KC;
      tdm_load_2d((unsigned int)(uintptr_t)&ldsA[buf][0],  hB + k2,     128);
      tdm_load_2d((unsigned int)(uintptr_t)&ldsBr[buf][0], wrow_r + k2, 64);
      tdm_load_2d((unsigned int)(uintptr_t)&ldsBz[buf][0], wrow_z + k2, 64);
    }
  }
#else
  __shared__ __bf16 ldsA[128 * LDSS];
  __shared__ __bf16 ldsBr[64 * LDSS];
  __shared__ __bf16 ldsBz[64 * LDSS];
  const int am = tid >> 1, ah = (tid & 1) * 16;
  const int bnr = tid >> 2, bq = (tid & 3) * 8;
  const size_t arow = (size_t)am * HID + ah;
  const size_t brow = (size_t)(bn * 64 + bnr) * HID + bq;
  for (int k0 = 0; k0 < HID; k0 += KC) {
    const uint4* ap = (const uint4*)(hB + arow + k0);
    __bf16* adst = &ldsA[am * LDSS + ah];
    ((uint4*)adst)[0] = ap[0];
    ((uint4*)adst)[1] = ap[1];
    *(uint4*)&ldsBr[bnr * LDSS + bq] = *(const uint4*)(Whr + brow + k0);
    *(uint4*)&ldsBz[bnr * LDSS + bq] = *(const uint4*)(Whz + brow + k0);
    __syncthreads();
    v16bf a0 = frag_a(ldsA, lane, wm * 32);
    v16bf a1 = frag_a(ldsA, lane, wm * 32 + 16);
    v16bf r0 = frag_b(ldsBr, lane, wn * 32);
    v16bf r1 = frag_b(ldsBr, lane, wn * 32 + 16);
    v16bf z0 = frag_b(ldsBz, lane, wn * 32);
    v16bf z1 = frag_b(ldsBz, lane, wn * 32 + 16);
    aR[0][0] = WMMA_BF16(a0, r0, aR[0][0]);
    aR[0][1] = WMMA_BF16(a0, r1, aR[0][1]);
    aR[1][0] = WMMA_BF16(a1, r0, aR[1][0]);
    aR[1][1] = WMMA_BF16(a1, r1, aR[1][1]);
    aZ[0][0] = WMMA_BF16(a0, z0, aZ[0][0]);
    aZ[0][1] = WMMA_BF16(a0, z1, aZ[0][1]);
    aZ[1][0] = WMMA_BF16(a1, z0, aZ[1][0]);
    aZ[1][1] = WMMA_BF16(a1, z1, aZ[1][1]);
    __syncthreads();
  }
#endif
  const int mloc = (lane >> 4) * 8, nloc = lane & 15;
#pragma unroll
  for (int ti = 0; ti < 2; ++ti)
#pragma unroll
    for (int tj = 0; tj < 2; ++tj) {
      int gn = bn * 64 + wn * 32 + tj * 16 + nloc;
#pragma unroll
      for (int j = 0; j < 8; ++j) {
        int gm = wm * 32 + ti * 16 + mloc + j;   // batch row, 0..127
        size_t idx = (size_t)gm * HID + gn;
        float R = sigmoid_f(aR[ti][tj][j] + Pr[idx]);
        float Z = sigmoid_f(aZ[ti][tj][j] + Pz[idx]);
        float ho = h[idx];
        Zbuf[idx] = Z;
        RhB[idx] = (__bf16)(R * ho);
      }
    }
}

// ---------------- per-step candidate + state update (M=128, N=HID, K=HID) --------------
__global__ __launch_bounds__(256)
void gate_h_kernel(const __bf16* __restrict__ RhB, const __bf16* __restrict__ Whh,
                   const float* __restrict__ Ph, const float* __restrict__ Zbuf,
                   float* __restrict__ h, __bf16* __restrict__ hB,
                   __bf16* __restrict__ HsB) {
  const int tid = threadIdx.x, lane = tid & 31, wave = tid >> 5;
  const int wm = wave & 3, wn = wave >> 2;
  const int bn = blockIdx.x;
  v8f acc[2][2] = {};

#if GRU_USE_TDM
  __shared__ __bf16 ldsA[2][128 * LDSS];
  __shared__ __bf16 ldsB[2][64 * LDSS];
  const __bf16* wrow = Whh + (size_t)(bn * 64) * HID;
  if (wave == 0) {
    tdm_load_2d((unsigned int)(uintptr_t)&ldsA[0][0], RhB,       128);
    tdm_load_2d((unsigned int)(uintptr_t)&ldsB[0][0], wrow,      64);
    tdm_load_2d((unsigned int)(uintptr_t)&ldsA[1][0], RhB + KC,  128);
    tdm_load_2d((unsigned int)(uintptr_t)&ldsB[1][0], wrow + KC, 64);
  }
  for (int i = 0; i < NCH; ++i) {
    if (i < NCH - 1) __builtin_amdgcn_s_wait_tensorcnt(2);
    else             __builtin_amdgcn_s_wait_tensorcnt(0);
    __syncthreads();
    const int buf = i & 1;
    v16bf a0 = frag_a(ldsA[buf], lane, wm * 32);
    v16bf a1 = frag_a(ldsA[buf], lane, wm * 32 + 16);
    v16bf b0 = frag_b(ldsB[buf], lane, wn * 32);
    v16bf b1 = frag_b(ldsB[buf], lane, wn * 32 + 16);
    acc[0][0] = WMMA_BF16(a0, b0, acc[0][0]);
    acc[0][1] = WMMA_BF16(a0, b1, acc[0][1]);
    acc[1][0] = WMMA_BF16(a1, b0, acc[1][0]);
    acc[1][1] = WMMA_BF16(a1, b1, acc[1][1]);
    __syncthreads();
    if (wave == 0 && i + 2 < NCH) {
      const int k2 = (i + 2) * KC;
      tdm_load_2d((unsigned int)(uintptr_t)&ldsA[buf][0], RhB + k2,  128);
      tdm_load_2d((unsigned int)(uintptr_t)&ldsB[buf][0], wrow + k2, 64);
    }
  }
#else
  __shared__ __bf16 ldsA[128 * LDSS];
  __shared__ __bf16 ldsB[64 * LDSS];
  const int am = tid >> 1, ah = (tid & 1) * 16;
  const int bnr = tid >> 2, bq = (tid & 3) * 8;
  const size_t arow = (size_t)am * HID + ah;
  const size_t brow = (size_t)(bn * 64 + bnr) * HID + bq;
  for (int k0 = 0; k0 < HID; k0 += KC) {
    const uint4* ap = (const uint4*)(RhB + arow + k0);
    __bf16* adst = &ldsA[am * LDSS + ah];
    ((uint4*)adst)[0] = ap[0];
    ((uint4*)adst)[1] = ap[1];
    *(uint4*)&ldsB[bnr * LDSS + bq] = *(const uint4*)(Whh + brow + k0);
    __syncthreads();
    v16bf a0 = frag_a(ldsA, lane, wm * 32);
    v16bf a1 = frag_a(ldsA, lane, wm * 32 + 16);
    v16bf b0 = frag_b(ldsB, lane, wn * 32);
    v16bf b1 = frag_b(ldsB, lane, wn * 32 + 16);
    acc[0][0] = WMMA_BF16(a0, b0, acc[0][0]);
    acc[0][1] = WMMA_BF16(a0, b1, acc[0][1]);
    acc[1][0] = WMMA_BF16(a1, b0, acc[1][0]);
    acc[1][1] = WMMA_BF16(a1, b1, acc[1][1]);
    __syncthreads();
  }
#endif
  const int mloc = (lane >> 4) * 8, nloc = lane & 15;
#pragma unroll
  for (int ti = 0; ti < 2; ++ti)
#pragma unroll
    for (int tj = 0; tj < 2; ++tj) {
      int gn = bn * 64 + wn * 32 + tj * 16 + nloc;
#pragma unroll
      for (int j = 0; j < 8; ++j) {
        int gm = wm * 32 + ti * 16 + mloc + j;
        size_t idx = (size_t)gm * HID + gn;
        float hh = tanhf(acc[ti][tj][j] + Ph[idx]);
        float Z = Zbuf[idx];
        float ho = h[idx];
        float hn = Z * ho + (1.0f - Z) * hh;   // each idx touched by exactly one lane
        h[idx] = hn;
        __bf16 hb = (__bf16)hn;
        hB[idx] = hb;
        HsB[idx] = hb;                          // HsB pre-offset by t on host
      }
    }
}

// ---------------- out = Hs @ Wout^T + bout (M=T*B, N=OUT_DIM, K=HID) -------------------
__global__ __launch_bounds__(256)
void out_proj_kernel(const __bf16* __restrict__ Hs, const __bf16* __restrict__ Wo,
                     const float* __restrict__ bout, float* __restrict__ out) {
  __shared__ __bf16 ldsA[128 * LDSS];
  __shared__ __bf16 ldsB[64 * LDSS];
  const int tid = threadIdx.x, lane = tid & 31, wave = tid >> 5;
  const int wm = wave & 3, wn = wave >> 2;
  const int bm = blockIdx.y, bn = blockIdx.x;

  v8f acc[2][2] = {};
  const int am = tid >> 1, ah = (tid & 1) * 16;
  const int bnr = tid >> 2, bq = (tid & 3) * 8;
  const size_t arow = (size_t)(bm * 128 + am) * HID + ah;
  const size_t brow = (size_t)(bn * 64 + bnr) * HID + bq;

  for (int k0 = 0; k0 < HID; k0 += KC) {
    const uint4* ap = (const uint4*)(Hs + arow + k0);
    __bf16* adst = &ldsA[am * LDSS + ah];
    ((uint4*)adst)[0] = ap[0];
    ((uint4*)adst)[1] = ap[1];
    *(uint4*)&ldsB[bnr * LDSS + bq] = *(const uint4*)(Wo + brow + k0);
    if (k0 + KC < HID) __builtin_prefetch(Hs + arow + k0 + KC, 0, 1);
    __syncthreads();
    v16bf a0 = frag_a(ldsA, lane, wm * 32);
    v16bf a1 = frag_a(ldsA, lane, wm * 32 + 16);
    v16bf b0 = frag_b(ldsB, lane, wn * 32);
    v16bf b1 = frag_b(ldsB, lane, wn * 32 + 16);
    acc[0][0] = WMMA_BF16(a0, b0, acc[0][0]);
    acc[0][1] = WMMA_BF16(a0, b1, acc[0][1]);
    acc[1][0] = WMMA_BF16(a1, b0, acc[1][0]);
    acc[1][1] = WMMA_BF16(a1, b1, acc[1][1]);
    __syncthreads();
  }
  const int mloc = (lane >> 4) * 8, nloc = lane & 15;
#pragma unroll
  for (int ti = 0; ti < 2; ++ti)
#pragma unroll
    for (int tj = 0; tj < 2; ++tj) {
      int gn = bn * 64 + wn * 32 + tj * 16 + nloc;
      float bb = bout[gn];
#pragma unroll
      for (int j = 0; j < 8; ++j) {
        int gm = bm * 128 + wm * 32 + ti * 16 + mloc + j;
        out[(size_t)gm * OUT_DIM + gn] = acc[ti][tj][j] + bb;
      }
    }
}

// ---------------- host orchestration ----------------
extern "C" void kernel_launch(void* const* d_in, const int* in_sizes, int n_in,
                              void* d_out, int out_size, void* d_ws, size_t ws_size,
                              hipStream_t stream) {
  const float* x    = (const float*)d_in[0];
  const float* rn   = (const float*)d_in[1];
  const float* zn   = (const float*)d_in[2];
  const float* hn   = (const float*)d_in[3];
  const float* Wxz  = (const float*)d_in[4];
  const float* Wxr  = (const float*)d_in[5];
  const float* Wxh  = (const float*)d_in[6];
  const float* Whz  = (const float*)d_in[7];
  const float* bz   = (const float*)d_in[8];
  const float* Whr  = (const float*)d_in[9];
  const float* br   = (const float*)d_in[10];
  const float* Whh  = (const float*)d_in[11];
  const float* bh   = (const float*)d_in[12];
  const float* Wout = (const float*)d_in[13];
  const float* bout = (const float*)d_in[14];
  float* out = (float*)d_out;

  char* wsp = (char*)d_ws;
  auto take = [&](size_t bytes) -> char* {
    char* p = wsp;
    wsp += (bytes + 255) & ~(size_t)255;
    return p;
  };
  const size_t TBH = (size_t)T_STEPS * BATCH * HID;
  __bf16* WxrB = (__bf16*)take((size_t)HID * IN_DIM * 2);
  __bf16* WxzB = (__bf16*)take((size_t)HID * IN_DIM * 2);
  __bf16* WxhB = (__bf16*)take((size_t)HID * IN_DIM * 2);
  __bf16* WhrB = (__bf16*)take((size_t)HID * HID * 2);
  __bf16* WhzB = (__bf16*)take((size_t)HID * HID * 2);
  __bf16* WhhB = (__bf16*)take((size_t)HID * HID * 2);
  __bf16* WoB  = (__bf16*)take((size_t)OUT_DIM * HID * 2);
  float*  Pr   = (float*)take(TBH * 4);
  float*  Pz   = (float*)take(TBH * 4);
  float*  Ph   = (float*)take(TBH * 4);
  float*  hbuf = (float*)take((size_t)BH * 4);
  float*  Zbuf = (float*)take((size_t)BH * 4);
  __bf16* hB   = (__bf16*)take((size_t)BH * 2);
  __bf16* RhB  = (__bf16*)take((size_t)BH * 2);
  __bf16* HsB  = (__bf16*)take(TBH * 2);

  // 1) weights -> bf16 (once per launch; deterministic)
  const int nWx = HID * IN_DIM, nWh = HID * HID, nWo = OUT_DIM * HID;
  cvt_f32_bf16<<<nWx / 1024, 256, 0, stream>>>(Wxr, WxrB, nWx);
  cvt_f32_bf16<<<nWx / 1024, 256, 0, stream>>>(Wxz, WxzB, nWx);
  cvt_f32_bf16<<<nWx / 1024, 256, 0, stream>>>(Wxh, WxhB, nWx);
  cvt_f32_bf16<<<nWh / 1024, 256, 0, stream>>>(Whr, WhrB, nWh);
  cvt_f32_bf16<<<nWh / 1024, 256, 0, stream>>>(Whz, WhzB, nWh);
  cvt_f32_bf16<<<nWh / 1024, 256, 0, stream>>>(Whh, WhhB, nWh);
  cvt_f32_bf16<<<nWo / 1024, 256, 0, stream>>>(Wout, WoB, nWo);
  init_h_kernel<<<(BH + 255) / 256, 256, 0, stream>>>(hbuf, hB, BH);

  // 2) time-parallel input projections (bias folded in)
  dim3 gP(HID / 64, (T_STEPS * BATCH) / 128);
  input_proj_kernel<<<gP, 256, 0, stream>>>(x, rn, WxrB, br, Pr, T_STEPS * BATCH, HID, IN_DIM);
  input_proj_kernel<<<gP, 256, 0, stream>>>(x, zn, WxzB, bz, Pz, T_STEPS * BATCH, HID, IN_DIM);
  input_proj_kernel<<<gP, 256, 0, stream>>>(x, hn, WxhB, bh, Ph, T_STEPS * BATCH, HID, IN_DIM);

  // 3) sequential recurrence; stream order provides the inter-step dependency
  dim3 gG(HID / 64, 1);
  for (int t = 0; t < T_STEPS; ++t) {
    size_t o = (size_t)t * BH;
    gate_rz_kernel<<<gG, 256, 0, stream>>>(hB, WhrB, WhzB, Pr + o, Pz + o, hbuf, Zbuf, RhB);
    gate_h_kernel<<<gG, 256, 0, stream>>>(RhB, WhhB, Ph + o, Zbuf, hbuf, hB, HsB + o);
  }

  // 4) time-parallel output projection
  dim3 gO(OUT_DIM / 64, (T_STEPS * BATCH) / 128);
  out_proj_kernel<<<gO, 256, 0, stream>>>(HsB, WoB, bout, out);
}
